// MultiHeadAttention_22849226015144
// MI455X (gfx1250) — matrix operand order repaired and checked by reading, not compiled, not run on hardware
//
#include <hip/hip_runtime.h>
#include <stdint.h>

// MI455X / gfx1250: wave32, WMMA 16x16x32 f16 (fp32 accum).
typedef __attribute__((ext_vector_type(16))) _Float16 v16h;
typedef __attribute__((ext_vector_type(8)))  float    v8f;

#define LOG2E 1.44269504088896f

union F16x16 {
    uint32_t       u32[8];
    unsigned short u16[16];
    uint4          q[2];
    v16h           v;
};

// Packed f32x2 -> f16x2 via v_cvt_pk_rtz_f16_f32 (single instruction).
__device__ inline uint32_t pk2f(float a, float b) {
    auto h = __builtin_amdgcn_cvt_pkrtz(a, b);
    union { decltype(h) v; uint32_t u; } r;
    r.v = h;
    return r.u;
}

__device__ inline uint4 pk8(float4 a, float4 b) {
    uint4 r; r.x = pk2f(a.x, a.y); r.y = pk2f(a.z, a.w);
    r.z = pk2f(b.x, b.y); r.w = pk2f(b.z, b.w); return r;
}

__device__ inline v8f wmma_f16(const F16x16& a, const F16x16& b, v8f c) {
    return __builtin_amdgcn_wmma_f32_16x16x32_f16(
        /*neg_a=*/false, a.v, /*neg_b=*/false, b.v,
        /*c_mod=*/(short)0, c, /*reuse_a=*/false, /*reuse_b=*/false);
}

// A-fragment (16x32 f16, ISA 7.12.2): lane L holds row M=L&15.
// lanes 0-15 hold K={kb..kb+7, kb+16..kb+23} with kb=0; lanes 16-31 kb=8.
// Both 8-half runs are contiguous -> two 16B LDS reads (ds_load_b128).
__device__ inline void load_a_frag(F16x16& f, const unsigned short* row, int half) {
    const unsigned short* p = row + half * 8;
    f.q[0] = *(const uint4*)(p);
    f.q[1] = *(const uint4*)(p + 16);
}

// B-fragment (32x16 f16): lane L holds col N=L&15; K={kb..kb+7, kb+16..kb+23}
// with kb = (L>=16 ? 8 : 0), matching the A-fragment. With the operand staged
// K-contiguous per N row (transposed LDS), this is two 16B reads.
__device__ inline void load_b_frag(F16x16& f, const unsigned short* rowT, int half) {
    const unsigned short* p = rowT + half * 8;
    f.q[0] = *(const uint4*)(p);
    f.q[1] = *(const uint4*)(p + 16);
}

__device__ inline float red_max16(float v) {
    v = fmaxf(v, __shfl_xor(v, 1, 32));
    v = fmaxf(v, __shfl_xor(v, 2, 32));
    v = fmaxf(v, __shfl_xor(v, 4, 32));
    v = fmaxf(v, __shfl_xor(v, 8, 32));
    return v;
}
__device__ inline float red_sum16(float v) {
    v += __shfl_xor(v, 1, 32);
    v += __shfl_xor(v, 2, 32);
    v += __shfl_xor(v, 4, 32);
    v += __shfl_xor(v, 8, 32);
    return v;
}

// ---------------------------------------------------------------------------
// Kernel 1: QKV GEMM.  C[8192,3072] = X[8192,1024] @ W[1024,3072] + b
// fp32 -> f16 fused into LDS staging (v_cvt_pk_rtz_f16_f32); W staged
// TRANSPOSED (N-major) so B-fragments are contiguous b128 reads. Q scaled by
// 1/32 (=1/sqrt(D)); Q/K/V scattered f16 into [B,H,S,64].
// Block tile 128x128, K-step 32, 8 waves (4 M x 2 N), 32x64 per wave.
// ---------------------------------------------------------------------------
__global__ __launch_bounds__(256) void qkv_gemm_kernel(
    const float* __restrict__ X, const float* __restrict__ W,
    const float* __restrict__ bias,
    unsigned short* __restrict__ Qb, unsigned short* __restrict__ Kb,
    unsigned short* __restrict__ Vb)
{
    __shared__ __align__(16) unsigned short As [128][40];  // M-major, 32 K (+8 pad)
    __shared__ __align__(16) unsigned short BsT[128][40];  // N-major, 32 K (+8 pad)

    const int tid  = threadIdx.x;
    const int lane = tid & 31;
    const int wave = tid >> 5;
    const int wm   = wave & 3;
    const int wn   = wave >> 2;
    const int half = lane >> 4;
    const int l15  = lane & 15;
    const int mbase = half * 8;

    const int m0 = blockIdx.y * 128;
    const int n0 = blockIdx.x * 128;

    v8f acc[2][4];
#pragma unroll
    for (int i = 0; i < 2; i++)
#pragma unroll
        for (int j = 0; j < 4; j++) acc[i][j] = (v8f){0,0,0,0,0,0,0,0};

    const int arow = tid >> 1;            // 0..127
    const int acol = (tid & 1) * 16;      // 0/16
    const int bn   = (tid & 63) * 2;      // n: 0..126 (pair of N rows)
    const int bk8  = (tid >> 6) * 8;      // k-oct: 0/8/16/24

    for (int kb = 0; kb < 1024; kb += 32) {
        // ---- stage A (X fp32 -> f16, row-major) ----
        {
            const float* src = X + (size_t)(m0 + arow) * 1024 + kb + acol;
            float4 a0 = ((const float4*)src)[0];
            float4 a1 = ((const float4*)src)[1];
            float4 a2 = ((const float4*)src)[2];
            float4 a3 = ((const float4*)src)[3];
            *(uint4*)&As[arow][acol]     = pk8(a0, a1);
            *(uint4*)&As[arow][acol + 8] = pk8(a2, a3);
        }
        // ---- stage B (W fp32 -> f16, TRANSPOSED to N-major) ----
        {
            const float* src = W + (size_t)(kb + bk8) * 3072 + n0 + bn;
            uint32_t ra[4], rb[4];
#pragma unroll
            for (int j = 0; j < 8; j += 2) {
                float2 f0 = *(const float2*)(src + (size_t)j * 3072);
                float2 f1 = *(const float2*)(src + (size_t)(j + 1) * 3072);
                ra[j >> 1] = pk2f(f0.x, f1.x);
                rb[j >> 1] = pk2f(f0.y, f1.y);
            }
            *(uint4*)&BsT[bn][bk8]     = (uint4){ra[0], ra[1], ra[2], ra[3]};
            *(uint4*)&BsT[bn + 1][bk8] = (uint4){rb[0], rb[1], rb[2], rb[3]};
        }
        if (kb + 32 < 1024) {
            __builtin_prefetch(X + (size_t)(m0 + arow) * 1024 + kb + 32 + acol, 0, 1);
            __builtin_prefetch(W + (size_t)(kb + 32 + bk8) * 3072 + n0 + bn, 0, 1);
        }
        __syncthreads();

        F16x16 af[2];
        load_a_frag(af[0], &As[wm * 32 +      l15][0], half);
        load_a_frag(af[1], &As[wm * 32 + 16 + l15][0], half);

#pragma unroll
        for (int nt = 0; nt < 4; nt++) {
            F16x16 bf;
            load_b_frag(bf, &BsT[wn * 64 + nt * 16 + l15][0], half);
            acc[0][nt] = wmma_f16(af[0], bf, acc[0][nt]);
            acc[1][nt] = wmma_f16(af[1], bf, acc[1][nt]);
        }
        __syncthreads();
    }

    // ---- epilogue: scatter Q/K/V f16 (row pairs packed), Q scaled 1/32 ----
#pragma unroll
    for (int nt = 0; nt < 4; nt++) {
        const int col   = n0 + wn * 64 + nt * 16 + l15;
        const float bv  = bias[col];
        const int which = col >> 10;           // 0=Q 1=K 2=V
        const int d     = col & 1023;
        const int hh    = d >> 6;
        const int hd    = d & 63;
        unsigned short* dst = (which == 0) ? Qb : ((which == 1) ? Kb : Vb);
        const float scl = (which == 0) ? 0.03125f : 1.0f;
#pragma unroll
        for (int mt = 0; mt < 2; mt++) {
#pragma unroll
            for (int r = 0; r < 8; r += 2) {
                const int row = m0 + wm * 32 + mt * 16 + mbase + r;
                const int bb  = row >> 11;
                const int ss  = row & 2047;
                const size_t idx = (((size_t)(bb * 16 + hh)) * 2048 + ss) * 64 + hd;
                const uint32_t pp = pk2f((acc[mt][nt][r]     + bv) * scl,
                                         (acc[mt][nt][r + 1] + bv) * scl);
                dst[idx]      = (unsigned short)pp;         // row r
                dst[idx + 64] = (unsigned short)(pp >> 16); // row r+1 (s+1)
            }
        }
    }
}

// ---------------------------------------------------------------------------
// Kernel 2: causal flash attention per (b,h). Q/K/V f16 [B,H,S,64].
// WG = 128 query rows (8 waves x 16 rows, private online-softmax state).
// K staged row-major (QK^T B-frags contiguous over hd); V staged TRANSPOSED
// (hd-major, v_perm_b32 packing) so PV B-frags are contiguous over keys.
// Output f16 [B,S,1024].
// ---------------------------------------------------------------------------
__global__ __launch_bounds__(256) void attn_kernel(
    const unsigned short* __restrict__ Qb,
    const unsigned short* __restrict__ Kb,
    const unsigned short* __restrict__ Vb,
    unsigned short* __restrict__ Ob)
{
    __shared__ __align__(16) unsigned short Ks [32][72];      // [key][hd]
    __shared__ __align__(16) unsigned short VsT[64][40];      // [hd][key]
    __shared__ __align__(16) unsigned short Ps [8][16][40];   // per-wave P tile

    const int tid  = threadIdx.x;
    const int lane = tid & 31;
    const int wave = tid >> 5;
    const int half = lane >> 4;
    const int l15  = lane & 15;
    const int mbase = half * 8;

    const int bh  = blockIdx.y;                      // b*16 + h
    const int q0  = blockIdx.x * 128 + wave * 16;
    const size_t base = (size_t)bh * 2048 * 64;

    // Q A-fragments (hd 0..31 and 32..63) straight from global f16.
    F16x16 qlo, qhi;
    {
        const unsigned short* qrow = Qb + base + (size_t)(q0 + l15) * 64 + half * 8;
        qlo.q[0] = *(const uint4*)(qrow);
        qlo.q[1] = *(const uint4*)(qrow + 16);
        qhi.q[0] = *(const uint4*)(qrow + 32);
        qhi.q[1] = *(const uint4*)(qrow + 48);
    }

    float mrow[8], lrow[8];
    v8f o[4];
#pragma unroll
    for (int r = 0; r < 8; r++) { mrow[r] = -__builtin_inff(); lrow[r] = 0.0f; }
#pragma unroll
    for (int t = 0; t < 4; t++) o[t] = (v8f){0,0,0,0,0,0,0,0};

    const int kkey = tid >> 3;            // 0..31  (K staging)
    const int kcol = (tid & 7) * 8;       // 0..56
    const int vhd  = (tid & 31) * 2;      // 0..62  (V transposed staging)
    const int vkq  = (tid >> 5) * 4;      // 0..28
    const int nkt  = blockIdx.x * 4 + 4;

    for (int kt = 0; kt < nkt; kt++) {
        const int k0 = kt * 32;
        // ---- stage K row-major ----
        {
            uint4 kq = *(const uint4*)(Kb + base + (size_t)(k0 + kkey) * 64 + kcol);
            *(uint4*)&Ks[kkey][kcol] = kq;
        }
        // ---- stage V transposed: [hd][key] (byte-perm half merge) ----
        {
            const unsigned short* vsrc = Vb + base + (size_t)(k0 + vkq) * 64 + vhd;
            uint32_t u0 = *(const uint32_t*)(vsrc);
            uint32_t u1 = *(const uint32_t*)(vsrc + 64);
            uint32_t u2 = *(const uint32_t*)(vsrc + 128);
            uint32_t u3 = *(const uint32_t*)(vsrc + 192);
            uint32_t lo0 = __builtin_amdgcn_perm(u1, u0, 0x05040100u);
            uint32_t hi0 = __builtin_amdgcn_perm(u1, u0, 0x07060302u);
            uint32_t lo1 = __builtin_amdgcn_perm(u3, u2, 0x05040100u);
            uint32_t hi1 = __builtin_amdgcn_perm(u3, u2, 0x07060302u);
            *(uint2*)&VsT[vhd][vkq]     = (uint2){lo0, lo1};
            *(uint2*)&VsT[vhd + 1][vkq] = (uint2){hi0, hi1};
        }
        __syncthreads();

        if (k0 <= q0 + 15) {                         // wave-uniform guard
            v8f s0 = (v8f){0,0,0,0,0,0,0,0};
            v8f s1 = (v8f){0,0,0,0,0,0,0,0};

            // B-frags of K^T: contraction dim = hd (contiguous in Ks rows).
            F16x16 b00, b01, b10, b11;
            load_b_frag(b00, &Ks[l15][0],       half);
            load_b_frag(b01, &Ks[l15][32],      half);
            load_b_frag(b10, &Ks[l15 + 16][0],  half);
            load_b_frag(b11, &Ks[l15 + 16][32], half);
            s0 = wmma_f16(qlo, b00, s0);
            s0 = wmma_f16(qhi, b01, s0);
            s1 = wmma_f16(qlo, b10, s1);
            s1 = wmma_f16(qhi, b11, s1);

            // ---- online softmax (per row, 16 cols per lane-group) ----
#pragma unroll
            for (int r = 0; r < 8; r++) {
                const int row  = q0 + mbase + r;
                const int key0 = k0 + l15;
                const int key1 = k0 + 16 + l15;
                float x0 = (key0 <= row) ? s0[r] : -__builtin_inff();
                float x1 = (key1 <= row) ? s1[r] : -__builtin_inff();
                float mx   = red_max16(fmaxf(x0, x1));
                float mnew = fmaxf(mrow[r], mx);
                float rescale = exp2f((mrow[r] - mnew) * LOG2E);
                float p0 = exp2f((x0 - mnew) * LOG2E);
                float p1 = exp2f((x1 - mnew) * LOG2E);
                float rs = red_sum16(p0 + p1);
                lrow[r] = lrow[r] * rescale + rs;
                mrow[r] = mnew;
#pragma unroll
                for (int t = 0; t < 4; t++) o[t][r] *= rescale;
                const uint32_t pp = pk2f(p0, p1);
                Ps[wave][mbase + r][l15]      = (unsigned short)pp;
                Ps[wave][mbase + r][16 + l15] = (unsigned short)(pp >> 16);
            }

            // P back as A-fragment (16x32) from per-wave LDS.
            F16x16 pf;
            load_a_frag(pf, &Ps[wave][l15][0], half);

            // PV: 4 hd-tiles; V B-frags contiguous over keys (VsT rows).
#pragma unroll
            for (int vt = 0; vt < 4; vt++) {
                F16x16 vf;
                load_b_frag(vf, &VsT[vt * 16 + l15][0], half);
                o[vt] = wmma_f16(pf, vf, o[vt]);
            }
        }
        __syncthreads();
    }

    // ---- normalize + store f16 (row pairs packed) to [B,S,1024] ----
    const int bb = bh >> 4;
    const int hh = bh & 15;
#pragma unroll
    for (int r = 0; r < 8; r += 2) {
        const float inv0 = 1.0f / lrow[r];
        const float inv1 = 1.0f / lrow[r + 1];
        const int row = q0 + mbase + r;
        const size_t ob = ((size_t)bb * 2048 + row) * 1024 + hh * 64;
#pragma unroll
        for (int vt = 0; vt < 4; vt++) {
            const uint32_t pp = pk2f(o[vt][r] * inv0, o[vt][r + 1] * inv1);
            Ob[ob + vt * 16 + l15]        = (unsigned short)pp;
            Ob[ob + 1024 + vt * 16 + l15] = (unsigned short)(pp >> 16);
        }
    }
}

// ---------------------------------------------------------------------------
// Kernel 3: output GEMM. out[8192,1024] = Attn_f16[8192,1024] @ W_out + b_out
// Same structure as kernel 1 (A already f16, W_out staged transposed).
// ---------------------------------------------------------------------------
__global__ __launch_bounds__(256) void out_gemm_kernel(
    const unsigned short* __restrict__ Ain, const float* __restrict__ W,
    const float* __restrict__ bias, float* __restrict__ out)
{
    __shared__ __align__(16) unsigned short As [128][40];
    __shared__ __align__(16) unsigned short BsT[128][40];

    const int tid  = threadIdx.x;
    const int lane = tid & 31;
    const int wave = tid >> 5;
    const int wm   = wave & 3;
    const int wn   = wave >> 2;
    const int half = lane >> 4;
    const int l15  = lane & 15;
    const int mbase = half * 8;

    const int m0 = blockIdx.y * 128;
    const int n0 = blockIdx.x * 128;

    v8f acc[2][4];
#pragma unroll
    for (int i = 0; i < 2; i++)
#pragma unroll
        for (int j = 0; j < 4; j++) acc[i][j] = (v8f){0,0,0,0,0,0,0,0};

    const int arow = tid >> 1;
    const int acol = (tid & 1) * 16;
    const int bn   = (tid & 63) * 2;
    const int bk8  = (tid >> 6) * 8;

    for (int kb = 0; kb < 1024; kb += 32) {
        {   // A already f16
            const uint4* src = (const uint4*)(Ain + (size_t)(m0 + arow) * 1024 + kb + acol);
            *(uint4*)&As[arow][acol]     = src[0];
            *(uint4*)&As[arow][acol + 8] = src[1];
        }
        {   // W_out fp32 -> f16, transposed to N-major
            const float* src = W + (size_t)(kb + bk8) * 1024 + n0 + bn;
            uint32_t ra[4], rb[4];
#pragma unroll
            for (int j = 0; j < 8; j += 2) {
                float2 f0 = *(const float2*)(src + (size_t)j * 1024);
                float2 f1 = *(const float2*)(src + (size_t)(j + 1) * 1024);
                ra[j >> 1] = pk2f(f0.x, f1.x);
                rb[j >> 1] = pk2f(f0.y, f1.y);
            }
            *(uint4*)&BsT[bn][bk8]     = (uint4){ra[0], ra[1], ra[2], ra[3]};
            *(uint4*)&BsT[bn + 1][bk8] = (uint4){rb[0], rb[1], rb[2], rb[3]};
        }
        if (kb + 32 < 1024) {
            __builtin_prefetch(Ain + (size_t)(m0 + arow) * 1024 + kb + 32 + acol, 0, 1);
            __builtin_prefetch(W + (size_t)(kb + 32 + bk8) * 1024 + n0 + bn, 0, 1);
        }
        __syncthreads();

        F16x16 af[2];
        load_a_frag(af[0], &As[wm * 32 +      l15][0], half);
        load_a_frag(af[1], &As[wm * 32 + 16 + l15][0], half);

#pragma unroll
        for (int nt = 0; nt < 4; nt++) {
            F16x16 bf;
            load_b_frag(bf, &BsT[wn * 64 + nt * 16 + l15][0], half);
            acc[0][nt] = wmma_f16(af[0], bf, acc[0][nt]);
            acc[1][nt] = wmma_f16(af[1], bf, acc[1][nt]);
        }
        __syncthreads();
    }

#pragma unroll
    for (int nt = 0; nt < 4; nt++) {
        const int col  = n0 + wn * 64 + nt * 16 + l15;
        const float bv = bias[col];
#pragma unroll
        for (int mt = 0; mt < 2; mt++) {
#pragma unroll
            for (int r = 0; r < 8; r++) {
                const int row = m0 + wm * 32 + mt * 16 + mbase + r;
                out[(size_t)row * 1024 + col] = acc[mt][nt][r] + bv;
            }
        }
    }
}

// ---------------------------------------------------------------------------
extern "C" void kernel_launch(void* const* d_in, const int* in_sizes, int n_in,
                              void* d_out, int out_size, void* d_ws, size_t ws_size,
                              hipStream_t stream) {
    (void)in_sizes; (void)n_in; (void)out_size; (void)ws_size;
    const float* X     = (const float*)d_in[0];   // [4,2048,1024]
    const float* W_qkv = (const float*)d_in[1];   // [1024,3072]
    const float* b_qkv = (const float*)d_in[2];   // [3072]
    const float* W_out = (const float*)d_in[3];   // [1024,1024]
    const float* b_out = (const float*)d_in[4];   // [1024]
    float* out = (float*)d_out;                   // [4,2048,1024]

    const size_t QKV_ELEMS = (size_t)4 * 16 * 2048 * 64;   // 8,388,608
    unsigned short* Qb = (unsigned short*)d_ws;            // f16 [B,H,S,64]
    unsigned short* Kb = Qb + QKV_ELEMS;
    unsigned short* Vb = Kb + QKV_ELEMS;
    unsigned short* Ob = Vb + QKV_ELEMS;                   // f16 [B,S,1024]

    qkv_gemm_kernel<<<dim3(3072 / 128, 8192 / 128), 256, 0, stream>>>(
        X, W_qkv, b_qkv, Qb, Kb, Vb);
    attn_kernel<<<dim3(2048 / 128, 4 * 16), 256, 0, stream>>>(Qb, Kb, Vb, Ob);
    out_gemm_kernel<<<dim3(1024 / 128, 8192 / 128), 256, 0, stream>>>(
        Ob, W_out, b_out, out);
}
